// NeighborCorrelator_50757923504662
// MI455X (gfx1250) — matrix operand order repaired and checked by reading, not compiled, not run on hardware
//
#include <hip/hip_runtime.h>

typedef __attribute__((ext_vector_type(16))) __bf16 v16bf;
typedef __attribute__((ext_vector_type(8)))  float  v8f;

#define B_   4
#define C_   256
#define H_   256
#define W_   256
#define KS   7
#define PAD_ 3
#define NOFF 49

union AV { unsigned int u[8]; v16bf v; };

__device__ __forceinline__ unsigned short f2bf(float f) {
    // round-to-nearest-even f32 -> bf16
    unsigned int u = __float_as_uint(f);
    u += 0x7FFFu + ((u >> 16) & 1u);
    return (unsigned short)(u >> 16);
}

// Pass 1: inverse channel L2 norm per pixel (F.normalize denominator, eps-clamped)
__global__ void __launch_bounds__(256) inv_norm_kernel(const float* __restrict__ t,
                                                       float* __restrict__ inv) {
    const int HW = H_ * W_;
    int p  = blockIdx.x * blockDim.x + threadIdx.x;   // exact: B*H*W threads
    int b  = p / HW;
    int hw = p - b * HW;
    const float* base = t + (size_t)b * C_ * HW + hw;
    float s = 0.f;
#pragma unroll 4
    for (int c = 0; c < C_; ++c) {
        float v = base[(size_t)c * HW];
        s += v * v;
    }
    inv[p] = 1.0f / fmaxf(sqrtf(s), 1e-12f);
}

// Pass 2: banded-Gram WMMA correlation. One wave32 per (b, h, 16-wide w block).
__global__ void __launch_bounds__(32) corr_wmma_kernel(const float* __restrict__ x,
                                                       const float* __restrict__ y,
                                                       const float* __restrict__ invx,
                                                       const float* __restrict__ invy,
                                                       float* __restrict__ out) {
    const int lane = threadIdx.x;
    const int m    = lane & 15;   // matrix row / N index
    const int half = lane >> 4;   // lane half selects K sub-range
    const int w0   = blockIdx.x * 16;
    const int h    = blockIdx.y;
    const int b    = blockIdx.z;
    const size_t HW = (size_t)H_ * W_;

    __shared__ unsigned short xs[16][C_];   // [m][c]   bf16 X tile (normalized)
    __shared__ unsigned short ys[32][C_];   // [j][c]   bf16 Y strip, j = w' - (w0-3)
    __shared__ float band[16][32];          // Gram band for one di

    // Per-lane invnorm for X column w0+m (same for both lane halves)
    const float myInvX = invx[((size_t)b * H_ + h) * W_ + w0 + m];

    // ---- Load + normalize + convert X tile: lane covers fixed mm == m ----
    const float* xb = x + (size_t)b * C_ * HW + (size_t)h * W_ + w0;
    for (int idx = lane; idx < C_ * 16; idx += 32) {
        int c  = idx >> 4;
        int mm = idx & 15;        // == m for this lane
        xs[mm][c] = f2bf(xb[(size_t)c * HW + mm] * myInvX);
    }
    __syncthreads();

    // ---- Build A-matrix register chunks (16x32 bf16 A layout, ISA 7.12.2) ----
    // VGPR v: K = (v>=4 ? 16 : 0) + half*8 + (v&3)*2, pair (K, K+1); lane M = lane&15
    AV areg[8];
#pragma unroll
    for (int kk = 0; kk < 8; ++kk) {
#pragma unroll
        for (int v = 0; v < 8; ++v) {
            int kb = ((v >= 4) ? 16 : 0) + half * 8 + (v & 3) * 2;
            areg[kk].u[v] = *(const unsigned int*)&xs[m][kk * 32 + kb];
        }
    }

    for (int di = 0; di < KS; ++di) {
        int hy = h + di - PAD_;
        size_t obase = (((size_t)b * NOFF + di * KS) * H_ + h) * W_ + w0;

        if (hy < 0 || hy >= H_) {           // padded rows -> zero dot products
            for (int t = lane; t < 112; t += 32) {
                int dj = t >> 4, mm = t & 15;
                out[obase + (size_t)dj * HW + mm] = 0.f;
            }
            continue;
        }

        __syncthreads();  // previous di's readers of ys/band are done

        // Per-lane invnorm for Y column w' = w0-3+lane (lane == j always)
        int wpl = w0 - 3 + lane;
        float myInvY = (wpl >= 0 && wpl < W_)
                       ? invy[((size_t)b * H_ + hy) * W_ + wpl] : 0.f;

        const float* ybase = y + (size_t)b * C_ * HW + (size_t)hy * W_;
        if (di < KS - 1 && hy + 1 < H_)
            __builtin_prefetch(ybase + W_, 0, 1);   // warm next di's row

        for (int idx = lane; idx < C_ * 32; idx += 32) {
            int c = idx >> 5;
            int j = idx & 31;                 // == lane
            int wp = w0 - 3 + j;
            float v = (wp >= 0 && wp < W_) ? ybase[(size_t)c * HW + wp] : 0.f;
            ys[j][c] = f2bf(v * myInvY);
        }
        __syncthreads();

        // ---- Two 16x16 output tiles covering the 22-column band ----
#pragma unroll
        for (int nt = 0; nt < 2; ++nt) {
            v8f acc = {};
#pragma unroll
            for (int kk = 0; kk < 8; ++kk) {
                AV breg;
                // B-matrix 32x16 bf16 layout: VGPR v holds K = (v>=4 ? 16 : 0) + half*8 + (v&3)*2 (pair), N = lane&15
#pragma unroll
                for (int v = 0; v < 8; ++v) {
                    int kb = ((v >= 4) ? 16 : 0) + half * 8 + (v & 3) * 2;
                    breg.u[v] = *(const unsigned int*)&ys[nt * 16 + m][kk * 32 + kb];
                }
                acc = __builtin_amdgcn_wmma_f32_16x16x32_bf16(
                        /*neg_a=*/false, areg[kk].v,
                        /*neg_b=*/false, breg.v,
                        /*c_mod=*/(short)0, acc,
                        /*reuse_a=*/false, /*reuse_b=*/false);
            }
            // C/D layout: VGPR r holds row M = r + 8*half, col N = lane&15
#pragma unroll
            for (int r = 0; r < 8; ++r)
                band[r + 8 * half][nt * 16 + m] = acc[r];
        }
        __syncthreads();

        // ---- Extract 7 diagonals: out[(di,dj), w0+mm] = band[mm][mm+dj] ----
        for (int t = lane; t < 112; t += 32) {
            int dj = t >> 4, mm = t & 15;
            out[obase + (size_t)dj * HW + mm] = band[mm][mm + dj];
        }
    }
}

extern "C" void kernel_launch(void* const* d_in, const int* in_sizes, int n_in,
                              void* d_out, int out_size, void* d_ws, size_t ws_size,
                              hipStream_t stream) {
    (void)in_sizes; (void)n_in; (void)out_size; (void)ws_size;
    const float* x = (const float*)d_in[0];
    const float* y = (const float*)d_in[1];
    float* out  = (float*)d_out;
    float* invx = (float*)d_ws;                       // [B*H*W]
    float* invy = invx + (size_t)B_ * H_ * W_;        // [B*H*W]  (2 MB total)

    dim3 nblk((B_ * H_ * W_) / 256);
    hipLaunchKernelGGL(inv_norm_kernel, nblk, dim3(256), 0, stream, x, invx);
    hipLaunchKernelGGL(inv_norm_kernel, nblk, dim3(256), 0, stream, y, invy);

    dim3 grid(W_ / 16, H_, B_);
    hipLaunchKernelGGL(corr_wmma_kernel, grid, dim3(32), 0, stream,
                       x, y, invx, invy, out);
}
